// Interaction_GraphConvolution_20590073217235
// MI455X (gfx1250) — matrix operand order repaired and checked by reading, not compiled, not run on hardware
//
#include <hip/hip_runtime.h>
#include <hip/hip_bf16.h>

typedef __attribute__((ext_vector_type(16))) _Float16 v16h;
typedef __attribute__((ext_vector_type(8)))  _Float16 v8h;
typedef __attribute__((ext_vector_type(4)))  _Float16 v4h;
typedef __attribute__((ext_vector_type(2)))  _Float16 v2h;
typedef __attribute__((ext_vector_type(8)))  float    v8f;

#define N_NODES 1024
#define F_IN    256
#define D_IN    128
#define D_OUT   64

// A-operand (16x32 f16): lane m (0-15): K 0-15; lane m+16: K 16-31.
// Per-lane: 16 contiguous halves (16B at p, 16B at p+16B).
__device__ __forceinline__ v16h load_a16(const _Float16* p) {
    const v8h lo = *(const v8h*)(p);
    const v8h hi = *(const v8h*)(p + 8);
    v16h r;
#pragma unroll
    for (int t = 0; t < 8; ++t) { r[t] = lo[t]; r[8 + t] = hi[t]; }
    return r;
}

// B-operand (32x16 f16): lane n (0-15): K 0-15; lane n+16: K 16-31.
// Per-lane: 16 contiguous halves (two 16B LDS loads).
__device__ __forceinline__ v16h load_b16(const _Float16* p) {
    const v8h lo = *(const v8h*)(p);
    const v8h hi = *(const v8h*)(p + 8);
    v16h r;
#pragma unroll
    for (int t = 0; t < 8; ++t) { r[t] = lo[t]; r[8 + t] = hi[t]; }
    return r;
}

// ---------------------------------------------------------------------------
// Kernel 1: wf = (nf @ lin_w.T + lin_b) @ weight   -> fp32 (1024 x 64)
// ---------------------------------------------------------------------------
__global__ __launch_bounds__(128) void prep_wf_kernel(
    const float* __restrict__ nf, const float* __restrict__ lin_w,
    const float* __restrict__ lin_b, const float* __restrict__ weight,
    float* __restrict__ wf)
{
    __shared__ float s_nf[F_IN];
    __shared__ float s_x[D_IN];
    const int i = blockIdx.x;
    const int t = threadIdx.x;

    s_nf[t]       = nf[i * F_IN + t];
    s_nf[t + 128] = nf[i * F_IN + 128 + t];
    __syncthreads();

    float acc = lin_b[t];
    const float* wr = lin_w + t * F_IN;
#pragma unroll 8
    for (int k = 0; k < F_IN; ++k) acc += s_nf[k] * wr[k];
    s_x[t] = acc;
    __syncthreads();

    if (t < D_OUT) {
        float a = 0.f;
#pragma unroll 8
        for (int d = 0; d < D_IN; ++d) a += s_x[d] * weight[d * D_OUT + t];
        wf[i * D_OUT + t] = a;
    }
}

// ---------------------------------------------------------------------------
// Kernel 2: sib fp32 -> f16 (small non-negative integers: exact in f16)
// ---------------------------------------------------------------------------
__global__ __launch_bounds__(256) void conv_sib_kernel(
    const float* __restrict__ sib, _Float16* __restrict__ sib16)
{
    const int idx  = blockIdx.x * blockDim.x + threadIdx.x;
    const int base = idx * 4;
    const float4 v = *(const float4*)(sib + base);
    v4h o;
    o[0] = (_Float16)v.x; o[1] = (_Float16)v.y;
    o[2] = (_Float16)v.z; o[3] = (_Float16)v.w;
    *(v4h*)(sib16 + base) = o;
}

// ---------------------------------------------------------------------------
// Kernel 3 (main): per node i,
//   y[j,f] = A[i,j] * wf[j,f];  out[i,f] = y[:,f]^T sib y[:,f] / nc[i]^2
// LDS holds y transposed + f-tile-interleaved: row (n*4 + ft) = column f=ft*16+n
// (f-tile stride = 2KB -> DS immediate offsets; no per-iter VALU addressing)
// grid = 1024 blocks, 256 threads = 8 waves
// ---------------------------------------------------------------------------
__global__ __launch_bounds__(256) void gconv_main_kernel(
    const float* __restrict__ A,        // adjacency (== mask_father)
    const float* __restrict__ nc,       // neighbor_count
    const float* __restrict__ wf,       // 1024 x 64 fp32
    const _Float16* __restrict__ sib16, // 1024 x 1024 f16 (== mask_hadamard)
    float* __restrict__ out)
{
    extern __shared__ char smem[];
    _Float16* yT     = (_Float16*)smem;                       // 64 rows x 1024 f16
    float*    outbuf = (float*)(smem + D_OUT * N_NODES * 2);  // 64 f32

    const int i   = blockIdx.x;
    const int tid = threadIdx.x;

    if (tid < D_OUT) outbuf[tid] = 0.f;

    // ---- build yT[(n*4+ft)][j] = A[i,j] * wf[j, ft*16+n] ----
    const float* Arow = A + i * N_NODES;
    for (int idx = tid; idx < D_OUT * (N_NODES / 2); idx += 256) {
        const int f  = idx >> 9;                       // 0..63 (column)
        const int jp = idx & 511;                      // 0..511
        const int j  = jp * 2;
        const int rf = ((f & 15) << 2) | (f >> 4);     // interleaved LDS row
        const float a0 = Arow[j];
        const float a1 = Arow[j + 1];
        const float w0 = wf[j * D_OUT + f];
        const float w1 = wf[(j + 1) * D_OUT + f];
        v2h p;
        p[0] = (_Float16)(a0 * w0);
        p[1] = (_Float16)(a1 * w1);
        *(v2h*)(&yT[rf * N_NODES + j]) = p;
    }
    __syncthreads();

    // ---- WMMA: Z = sib @ y, fold diag(y^T Z) ----
    const int wave = tid >> 5;
    const int lane = tid & 31;
    const int n    = lane & 15;   // tile column
    const int hi   = lane >> 4;   // half-wave select

    float facc[4] = {0.f, 0.f, 0.f, 0.f};

    for (int t = 0; t < 8; ++t) {
        // compiler-only barrier: forbid hoisting the (t-invariant) LDS
        // B-operand loads out of this loop (caused scratch spills otherwise)
        asm volatile("" ::: "memory");

        const int b0 = (wave * 8 + t) * 16;   // b-tile base row
        v8f acc[4] = {v8f{}, v8f{}, v8f{}, v8f{}};

        const _Float16* sA = sib16 + (b0 + n) * N_NODES + hi * 16;
        const _Float16* yB = yT + (n * 4) * N_NODES + hi * 16;

#pragma unroll 2
        for (int j0 = 0; j0 < N_NODES; j0 += 32) {
            const v16h av = load_a16(sA + j0);
            const v16h b0v = load_b16(yB + 0 * N_NODES + j0);
            const v16h b1v = load_b16(yB + 1 * N_NODES + j0);
            const v16h b2v = load_b16(yB + 2 * N_NODES + j0);
            const v16h b3v = load_b16(yB + 3 * N_NODES + j0);

            acc[0] = __builtin_amdgcn_wmma_f32_16x16x32_f16(false, av, false, b0v, (short)0, acc[0], false, false);
            acc[1] = __builtin_amdgcn_wmma_f32_16x16x32_f16(false, av, false, b1v, (short)0, acc[1], false, false);
            acc[2] = __builtin_amdgcn_wmma_f32_16x16x32_f16(false, av, false, b2v, (short)0, acc[2], false, false);
            acc[3] = __builtin_amdgcn_wmma_f32_16x16x32_f16(false, av, false, b3v, (short)0, acc[3], false, false);
        }

        // fold: facc[ft] += sum_r y[b0 + r + 8*hi, ft*16+n] * Z[r + 8*hi, n]
        const _Float16* yF = yT + (n * 4) * N_NODES + b0 + hi * 8;
#pragma unroll
        for (int ft = 0; ft < 4; ++ft) {
            const v8h yv = *(const v8h*)(yF + ft * N_NODES);
            float s = 0.f;
#pragma unroll
            for (int r = 0; r < 8; ++r) s += (float)yv[r] * acc[ft][r];
            facc[ft] += s;
        }
    }

#pragma unroll
    for (int ft = 0; ft < 4; ++ft)
        atomicAdd(&outbuf[ft * 16 + n], facc[ft]);   // ds_add_f32
    __syncthreads();

    if (tid < D_OUT) {
        const float c = nc[i];
        out[i * D_OUT + tid] = outbuf[tid] / (c * c);
    }
}

// ---------------------------------------------------------------------------
extern "C" void kernel_launch(void* const* d_in, const int* in_sizes, int n_in,
                              void* d_out, int out_size, void* d_ws, size_t ws_size,
                              hipStream_t stream) {
    (void)in_sizes; (void)n_in; (void)out_size; (void)ws_size;

    const float* nf    = (const float*)d_in[0]; // node_features    (1024,256)
    const float* A     = (const float*)d_in[1]; // adjacency_matrix (1024,1024)
    // d_in[2] = mask_father == A[:,None,:], redundant
    const float* nc    = (const float*)d_in[3]; // neighbor_count   (1024,1)
    const float* sib   = (const float*)d_in[4]; // mask_hadamard    (1024,1,1024)
    const float* lin_w = (const float*)d_in[5]; // (128,256)
    const float* lin_b = (const float*)d_in[6]; // (128,)
    const float* wgt   = (const float*)d_in[7]; // (128,64)
    float* out = (float*)d_out;

    // workspace layout
    float*    wf    = (float*)d_ws;                                   // 256 KB
    _Float16* sib16 = (_Float16*)((char*)d_ws + N_NODES * D_OUT * 4); // 2 MB

    const size_t lds_bytes = (size_t)D_OUT * N_NODES * 2 + D_OUT * 4; // 128 KB + 256 B
    (void)hipFuncSetAttribute((const void*)gconv_main_kernel,
                              hipFuncAttributeMaxDynamicSharedMemorySize,
                              (int)lds_bytes);

    prep_wf_kernel<<<N_NODES, 128, 0, stream>>>(nf, lin_w, lin_b, wgt, wf);
    conv_sib_kernel<<<(N_NODES * N_NODES) / (256 * 4), 256, 0, stream>>>(sib, sib16);
    gconv_main_kernel<<<N_NODES, 256, lds_bytes, stream>>>(A, nc, wf, sib16, out);
}
